// LlamaAttention_1949915152492
// MI455X (gfx1250) — matrix operand order repaired and checked by reading, not compiled, not run on hardware
//
#include <hip/hip_runtime.h>
#include <stdint.h>

#define AS1 __attribute__((address_space(1)))
#define AS3 __attribute__((address_space(3)))

typedef __attribute__((ext_vector_type(16))) __bf16        v16bf;
typedef __attribute__((ext_vector_type(2)))  __bf16        v2bf;
typedef __attribute__((ext_vector_type(8)))  float         v8f;
typedef __attribute__((ext_vector_type(8)))  unsigned int  v8u;
typedef int v4i __attribute__((vector_size(16)));   // matches builtin's int4 pointee

#if defined(__AMDGCN__) && __has_builtin(__builtin_amdgcn_global_load_async_to_lds_b128) && __has_builtin(__builtin_amdgcn_s_wait_asynccnt)
#define HAS_ASYNC_LDS 1
#else
#define HAS_ASYNC_LDS 0
#endif

static __device__ __forceinline__ v8f wmma_bf16f32(v16bf a, v16bf b, v8f c) {
  // D = A(16x32 bf16) * B(32x16 bf16) + C(16x16 f32)
  return __builtin_amdgcn_wmma_f32_16x16x32_bf16(false, a, false, b, (short)0, c, false, false);
}
static __device__ __forceinline__ v8f v8f_zero() {
  v8f z = {0.f, 0.f, 0.f, 0.f, 0.f, 0.f, 0.f, 0.f};
  return z;
}

// ---------------- mean(|W|): deterministic two-pass reduction ----------------
__global__ void abssum_partial(const float* __restrict__ w, long long n, float* __restrict__ partial) {
  __shared__ float red[256];
  long long stride = (long long)gridDim.x * blockDim.x;
  float s = 0.f;
  for (long long i = (long long)blockIdx.x * blockDim.x + threadIdx.x; i < n; i += stride)
    s += fabsf(w[i]);
  red[threadIdx.x] = s;
  __syncthreads();
  for (int off = 128; off > 0; off >>= 1) {
    if ((int)threadIdx.x < off) red[threadIdx.x] += red[threadIdx.x + off];
    __syncthreads();
  }
  if (threadIdx.x == 0) partial[blockIdx.x] = red[0];
}

__global__ void abssum_final(const float* __restrict__ partial, float inv_n, float* __restrict__ scale_out) {
  __shared__ float red[256];
  red[threadIdx.x] = partial[threadIdx.x];
  __syncthreads();
  for (int off = 128; off > 0; off >>= 1) {
    if ((int)threadIdx.x < off) red[threadIdx.x] += red[threadIdx.x + off];
    __syncthreads();
  }
  if (threadIdx.x == 0) *scale_out = red[0] * inv_n;   // mean(|W|)
}

// ------------- Wq = round(W / (scale+eps)) as bf16 (exact small ints) --------
__global__ void quantize_w(const float* __restrict__ w, __bf16* __restrict__ wq,
                           long long n, const float* __restrict__ scale_p) {
  float inv = 1.f / (*scale_p + 1e-5f);
  long long stride = (long long)gridDim.x * blockDim.x;
  for (long long i = (long long)blockIdx.x * blockDim.x + threadIdx.x; i < n; i += stride)
    wq[i] = (__bf16)rintf(w[i] * inv);
}

__global__ void to_bf16(const float* __restrict__ x, __bf16* __restrict__ xb, long long n) {
  long long stride = (long long)gridDim.x * blockDim.x;
  for (long long i = (long long)blockIdx.x * blockDim.x + threadIdx.x; i < n; i += stride)
    xb[i] = (__bf16)x[i];
}

// ---------------- WMMA GEMM: out[M,N] = A[M,K] * Bw[N,K]^T * scale ----------
// Block: 256 thr = 8 waves; block tile 128x128; wave tile 32x64 (2x4 WMMA accs).
template <bool OUT_F32>
__global__ __launch_bounds__(256) void gemm_bf16_tn(
    const __bf16* __restrict__ A, const __bf16* __restrict__ Bw,
    void* __restrict__ outp, int M, int N, int K, const float* __restrict__ scale_p) {
  __shared__ unsigned int As[128][20];   // 128 rows x 32 bf16 (16 uints) + pad
  __shared__ unsigned int Bs[128][20];

  const int tid = threadIdx.x;
  const int lane = tid & 31;
  const int wv = tid >> 5;
  const int l15 = lane & 15;
  const int kh = lane >> 4;
  const int wm = (wv >> 1) * 32;
  const int wn = (wv & 1) * 64;
  const int m0 = blockIdx.y * 128;
  const int n0 = blockIdx.x * 128;

  v8f acc[2][4];
#pragma unroll
  for (int i = 0; i < 2; ++i)
#pragma unroll
    for (int j = 0; j < 4; ++j) acc[i][j] = v8f_zero();

  for (int k0 = 0; k0 < K; k0 += 32) {
    __syncthreads();
#pragma unroll
    for (int cc = 0; cc < 2; ++cc) {
      int ch = tid + cc * 256;       // 512 x 16B chunks per tile
      int row = ch >> 2;
      int c4 = ch & 3;
      const uint4* ga = (const uint4*)(A  + (long long)(m0 + row) * K + k0 + c4 * 8);
      const uint4* gb = (const uint4*)(Bw + (long long)(n0 + row) * K + k0 + c4 * 8);
#if HAS_ASYNC_LDS
      __builtin_amdgcn_global_load_async_to_lds_b128((AS1 v4i*)(void*)ga, (AS3 v4i*)(void*)&As[row][c4 * 4], 0, 0);
      __builtin_amdgcn_global_load_async_to_lds_b128((AS1 v4i*)(void*)gb, (AS3 v4i*)(void*)&Bs[row][c4 * 4], 0, 0);
#else
      *(uint4*)&As[row][c4 * 4] = *ga;
      *(uint4*)&Bs[row][c4 * 4] = *gb;
#endif
    }
#if HAS_ASYNC_LDS
    __builtin_amdgcn_s_wait_asynccnt(0);
#endif
    __syncthreads();

    v16bf af[2];
#pragma unroll
    for (int t = 0; t < 2; ++t) {    // A frag: k0/2 = v + 8*lanehalf
      v8u u;
#pragma unroll
      for (int v = 0; v < 8; ++v)
        u[v] = As[wm + t * 16 + l15][v + 8 * kh];
      af[t] = __builtin_bit_cast(v16bf, u);
    }
    v16bf bfb[4];
#pragma unroll
    for (int t = 0; t < 4; ++t) {    // B frag: k0/2 = v + 8*lanehalf
      v8u u;
#pragma unroll
      for (int v = 0; v < 8; ++v)
        u[v] = Bs[wn + t * 16 + l15][v + 8 * kh];
      bfb[t] = __builtin_bit_cast(v16bf, u);
    }
#pragma unroll
    for (int i = 0; i < 2; ++i)
#pragma unroll
      for (int j = 0; j < 4; ++j)
        acc[i][j] = wmma_bf16f32(af[i], bfb[j], acc[i][j]);
  }

  const float scale = *scale_p;
#pragma unroll
  for (int i = 0; i < 2; ++i)
#pragma unroll
    for (int j = 0; j < 4; ++j)
#pragma unroll
      for (int r = 0; r < 8; ++r) {
        int row = m0 + wm + i * 16 + r + 8 * kh;   // C layout: M = r + 8*lanehalf
        int col = n0 + wn + j * 16 + l15;          // N = lane&15
        float val = acc[i][j][r] * scale;
        if (OUT_F32) ((float*)outp)[(long long)row * N + col] = val;
        else         ((__bf16*)outp)[(long long)row * N + col] = (__bf16)val;
      }
}

// ---------------- RoPE in-place on bf16 qkv buffer [4096, 6144] -------------
__global__ void rope_inplace(__bf16* __restrict__ qkv, const float* __restrict__ rotary) {
  int tid = blockIdx.x * 256 + threadIdx.x;   // 4096*2*16*64 = 8,388,608 threads
  int p  = tid & 63;
  int h  = (tid >> 6) & 15;
  int qk = (tid >> 10) & 1;                   // 0=q, 1=k
  int m  = tid >> 11;                         // row in [0,4096)
  int s  = m & 2047;
  long long base = (long long)m * 6144 + qk * 2048 + h * 128;
  float x1 = (float)qkv[base + p];
  float x2 = (float)qkv[base + p + 64];
  const float* sinp = rotary + (long long)s * 128;                 // rotary[0]
  const float* cosp = rotary + 2048LL * 128 + (long long)s * 128;  // rotary[1]
  qkv[base + p]      = (__bf16)(x1 * cosp[p]      - x2 * sinp[p]);
  qkv[base + p + 64] = (__bf16)(x2 * cosp[p + 64] + x1 * sinp[p + 64]);
}

// ---------------- causal flash attention, bf16 WMMA, f32 softmax ------------
// grid: (16 q-blocks of 128 rows, 32 (b,h)); block: 256 thr = 8 waves, 16 q rows/wave
__global__ __launch_bounds__(256) void flash_attn_causal(
    const __bf16* __restrict__ qkv, __bf16* __restrict__ y) {
  __shared__ unsigned int Qs[128][68];    // 128 q rows x 128 d (bf16)
  __shared__ unsigned int Ks[32][68];     // 32 keys  x 128 d
  __shared__ unsigned int VTu[128][20];   // transposed: [d][key], pairs along key
  __shared__ unsigned int Pu[8][16][17];  // per-wave P staging (16 q x 32 k)

  const int tid = threadIdx.x;
  const int lane = tid & 31;
  const int wv = tid >> 5;
  const int l15 = lane & 15;
  const int kh = lane >> 4;
  const int qb = blockIdx.x;
  const int b = blockIdx.y >> 4;
  const int h = blockIdx.y & 15;
  const long long base_bs = (long long)b * 2048 * 6144;

#pragma unroll
  for (int cc = 0; cc < 8; ++cc) {        // load Q tile
    int ch = tid + cc * 256;
    int row = ch >> 4, c4 = ch & 15;
    const uint4* g = (const uint4*)(qkv + base_bs + (long long)(qb * 128 + row) * 6144 + h * 128 + c4 * 8);
    *(uint4*)&Qs[row][c4 * 4] = *g;
  }

  v8f O[8];
#pragma unroll
  for (int n = 0; n < 8; ++n) O[n] = v8f_zero();
  float mrow[8], lrow[8];
#pragma unroll
  for (int r = 0; r < 8; ++r) { mrow[r] = -1e30f; lrow[r] = 0.f; }

  const float sm = 0.08838834764831845f;  // 1/sqrt(128)
  const int kvEnd = qb * 128 + 128;

  for (int kv0 = 0; kv0 < kvEnd; kv0 += 32) {
    __syncthreads();
#pragma unroll
    for (int cc = 0; cc < 2; ++cc) {      // K tile
      int ch = tid + cc * 256;
      int row = ch >> 4, c4 = ch & 15;
      const uint4* g = (const uint4*)(qkv + base_bs + (long long)(kv0 + row) * 6144 + 2048 + h * 128 + c4 * 8);
      *(uint4*)&Ks[row][c4 * 4] = *g;
    }
    __bf16* VTb = (__bf16*)VTu;           // V tile, transposed store
#pragma unroll
    for (int cc = 0; cc < 8; ++cc) {
      int u = tid + cc * 256;             // 2048 uints = 32 keys x 64 d-pairs
      int key = u >> 6, du = u & 63, d = du * 2;
      unsigned int val = *(const unsigned int*)(qkv + base_bs + (long long)(kv0 + key) * 6144 + 4096 + h * 128 + d);
      v2bf pr = __builtin_bit_cast(v2bf, val);
      VTb[d * 40 + key] = pr.x;
      VTb[(d + 1) * 40 + key] = pr.y;
    }
    __syncthreads();

    v8f s0 = v8f_zero(), s1 = v8f_zero(); // scores: this wave's 16 q x 32 keys
#pragma unroll
    for (int d0 = 0; d0 < 128; d0 += 32) {
      v16bf aq;
      { v8u u;
#pragma unroll
        for (int v = 0; v < 8; ++v)
          u[v] = Qs[wv * 16 + l15][d0 / 2 + v + 8 * kh];
        aq = __builtin_bit_cast(v16bf, u); }
      v16bf bk0, bk1;
      { v8u u;
#pragma unroll
        for (int v = 0; v < 8; ++v) u[v] = Ks[l15][d0 / 2 + v + 8 * kh];
        bk0 = __builtin_bit_cast(v16bf, u); }
      { v8u u;
#pragma unroll
        for (int v = 0; v < 8; ++v) u[v] = Ks[16 + l15][d0 / 2 + v + 8 * kh];
        bk1 = __builtin_bit_cast(v16bf, u); }
      s0 = wmma_bf16f32(aq, bk0, s0);
      s1 = wmma_bf16f32(aq, bk1, s1);
    }

    const int qrow0 = qb * 128 + wv * 16 + 8 * kh;
#pragma unroll
    for (int r = 0; r < 8; ++r) {         // mask + online softmax per row
      int q = qrow0 + r;
      float a0 = s0[r] * sm, a1 = s1[r] * sm;
      if (kv0 + l15 > q)      a0 = -1e30f;
      if (kv0 + 16 + l15 > q) a1 = -1e30f;
      float vmax = fmaxf(a0, a1);
#pragma unroll
      for (int off = 1; off < 16; off <<= 1) vmax = fmaxf(vmax, __shfl_xor(vmax, off, 32));
      float mn = fmaxf(mrow[r], vmax);
      float alpha = __expf(mrow[r] - mn);
      mrow[r] = mn;
      float p0 = __expf(a0 - mn), p1 = __expf(a1 - mn);
      float rs = p0 + p1;
#pragma unroll
      for (int off = 1; off < 16; off <<= 1) rs += __shfl_xor(rs, off, 32);
      lrow[r] = lrow[r] * alpha + rs;
#pragma unroll
      for (int n = 0; n < 8; ++n) O[n][r] *= alpha;
      s0[r] = p0; s1[r] = p1;
    }

    __bf16* Pb = (__bf16*)&Pu[wv][0][0];  // stage P (C-layout -> A-layout via LDS)
#pragma unroll
    for (int r = 0; r < 8; ++r) {
      Pb[(r + 8 * kh) * 34 + l15]      = (__bf16)s0[r];
      Pb[(r + 8 * kh) * 34 + 16 + l15] = (__bf16)s1[r];
    }
    __syncthreads();

    v16bf ap;
    { v8u u;
#pragma unroll
      for (int v = 0; v < 8; ++v)
        u[v] = Pu[wv][l15][v + 8 * kh];
      ap = __builtin_bit_cast(v16bf, u); }
#pragma unroll
    for (int n = 0; n < 8; ++n) {         // O += P * V over 8 d-tiles
      v8u u;
#pragma unroll
      for (int v = 0; v < 8; ++v) u[v] = VTu[n * 16 + l15][v + 8 * kh];
      v16bf bv = __builtin_bit_cast(v16bf, u);
      O[n] = wmma_bf16f32(ap, bv, O[n]);
    }
  }

#pragma unroll
  for (int r = 0; r < 8; ++r) {
    float invl = 1.f / lrow[r];
    long long srow = (long long)qb * 128 + wv * 16 + r + 8 * kh;
#pragma unroll
    for (int n = 0; n < 8; ++n)
      y[((long long)b * 2048 + srow) * 2048 + h * 128 + n * 16 + l15] = (__bf16)(O[n][r] * invl);
  }
}

// ------------------------------- launch -------------------------------------
extern "C" void kernel_launch(void* const* d_in, const int* in_sizes, int n_in,
                              void* d_out, int out_size, void* d_ws, size_t ws_size,
                              hipStream_t stream) {
  (void)in_sizes; (void)n_in; (void)out_size; (void)ws_size;
  const float* x      = (const float*)d_in[0];   // [2,2048,2048]
  const float* rotary = (const float*)d_in[1];   // [2,2048,128]
  const float* qkvw   = (const float*)d_in[2];   // [6144,2048]
  const float* ow     = (const float*)d_in[3];   // [2048,2048]
  float* out = (float*)d_out;                    // [2,2048,2048] f32

  char* ws = (char*)d_ws;
  size_t off = 0;
  auto alloc = [&](size_t bytes) -> void* {
    void* p = ws + off;
    off = (off + bytes + 255) & ~(size_t)255;
    return p;
  };
  float*  partialA = (float*)alloc(256 * 4);
  float*  partialB = (float*)alloc(256 * 4);
  float*  scales   = (float*)alloc(2 * 4);
  __bf16* wq_qkv   = (__bf16*)alloc(6144ULL * 2048 * 2);
  __bf16* wq_o     = (__bf16*)alloc(2048ULL * 2048 * 2);
  __bf16* xb       = (__bf16*)alloc(4096ULL * 2048 * 2);
  __bf16* qkvb     = (__bf16*)alloc(4096ULL * 6144 * 2);
  __bf16* yb       = (__bf16*)alloc(4096ULL * 2048 * 2);

  abssum_partial<<<256, 256, 0, stream>>>(qkvw, 6144LL * 2048, partialA);
  abssum_partial<<<256, 256, 0, stream>>>(ow,   2048LL * 2048, partialB);
  abssum_final<<<1, 256, 0, stream>>>(partialA, 1.f / (6144.f * 2048.f), scales + 0);
  abssum_final<<<1, 256, 0, stream>>>(partialB, 1.f / (2048.f * 2048.f), scales + 1);

  quantize_w<<<2048, 256, 0, stream>>>(qkvw, wq_qkv, 6144LL * 2048, scales + 0);
  quantize_w<<<2048, 256, 0, stream>>>(ow,   wq_o,   2048LL * 2048, scales + 1);
  to_bf16<<<2048, 256, 0, stream>>>(x, xb, 4096LL * 2048);

  gemm_bf16_tn<false><<<dim3(48, 32), 256, 0, stream>>>(xb, wq_qkv, qkvb, 4096, 6144, 2048, scales + 0);
  rope_inplace<<<32768, 256, 0, stream>>>(qkvb, rotary);
  flash_attn_causal<<<dim3(16, 32), 256, 0, stream>>>(qkvb, yb);
  gemm_bf16_tn<true><<<dim3(16, 32), 256, 0, stream>>>(yb, wq_o, out, 4096, 2048, 2048, scales + 1);
}
